// MeshUnpool_34256659153543
// MI455X (gfx1250) — matrix operand order repaired and checked by reading, not compiled, not run on hardware
//
#include <hip/hip_runtime.h>
#include <hip/hip_bf16.h>

// MeshUnpool: out[b] = features[b] @ (unroll_mat[b] / occ[b][None,:])
//   features  : [4, 256, 4560]  f32
//   unroll_mat: [4, 4560, 9120] f32 (binary 0/1 -> exact in f16)
//   occ       : [4, 9120]       f32 (applied in fp32 at epilogue)
//   out       : [4, 256, 9120]  f32
//
// Memory-bound GEMM (~721 MB traffic, ~31us @ 23.3 TB/s).
// V_WMMA_F32_16X16X32_F16, fp32 accumulation, double-buffered LDS,
// one workgroup barrier per 64-K slab (12 WMMAs per wave per barrier).

typedef _Float16 v16h __attribute__((ext_vector_type(16)));
typedef _Float16 v8h  __attribute__((ext_vector_type(8)));
typedef float    v8f  __attribute__((ext_vector_type(8)));
typedef float    v4f  __attribute__((ext_vector_type(4)));

#define NFEAT   256
#define EDGES   4560
#define TARGETN 9120
#define BATCH   4

#define BM 128          // block tile M (256 = 2*128)
#define BN 96           // block tile N (9120 = 95*96)
#define BK 64           // K per staging round = 2 WMMA K-slabs
#define LSTRIDE 72      // halves per LDS row (64 + 8 pad; 144 B = 9*16 B)
#define NSTEP ((EDGES + BK - 1) / BK)   // 72 (last step 3/4 zero-filled)

union FragAB { v16h v; v8h h[2]; };

__global__ __launch_bounds__(256) void mesh_unpool_wmma(
    const float* __restrict__ features,
    const float* __restrict__ unroll_mat,
    const float* __restrict__ occ,
    float* __restrict__ out)
{
  // Double-buffered f16 tiles: A[M][K], B transposed [N][K].
  // 2*(128*72 + 96*72)*2B = 63 KB.
  __shared__ __align__(16) _Float16 As[2][BM * LSTRIDE];
  __shared__ __align__(16) _Float16 Bs[2][BN * LSTRIDE];

  const int tid  = threadIdx.x;
  const int lane = tid & 31;
  const int wave = tid >> 5;          // 8 waves
  const int lm   = lane & 15;         // row/col within a 16x16 tile
  const int lg   = lane >> 4;         // lane group (0/1)

  const int b     = blockIdx.z;
  const int Mbase = blockIdx.y * BM;
  const int Nbase = blockIdx.x * BN;

  const int wm = wave & 3;            // 4 M-groups of 32 rows
  const int wn = wave >> 2;           // 2 N-groups of 48 cols

  // ---- staging thread mapping (256 threads per BK=64 round) ----
  // A tile: 128 rows x 64 K fp32 -> each thread: 32 K of one row (8 x b128)
  const int arow  = tid >> 1;
  const int akoff = (tid & 1) * 32;
  // B tile: 64 K x 96 N fp32 -> each thread: 24 N of one K row (6 x b128)
  const int bkk = tid >> 2;           // 0..63
  const int bn0 = (tid & 3) * 24;     // 0,24,48,72

  const float* fptr = features   + ((size_t)b * NFEAT + (Mbase + arow)) * EDGES + akoff;
  const float* uptr = unroll_mat + ((size_t)b * EDGES + bkk) * TARGETN + (Nbase + bn0);

  const v4f zero4 = {0.f, 0.f, 0.f, 0.f};
  v4f areg[8], breg[6];

  auto load_regs = [&](int kbase) {
#pragma unroll
    for (int j = 0; j < 8; ++j) {
      const int k = kbase + akoff + 4 * j;
      areg[j] = (k < EDGES) ? *(const v4f*)(fptr + (size_t)kbase + 4 * j) : zero4;
    }
    const bool kok = (kbase + bkk) < EDGES;
#pragma unroll
    for (int j = 0; j < 6; ++j) {
      breg[j] = kok ? *(const v4f*)(uptr + (size_t)kbase * TARGETN + 4 * j) : zero4;
    }
  };

  auto store_lds = [&](int buf) {
    // A: cvt 32 f32 -> 32 f16, four aligned b128 LDS stores (row-major [M][K])
    union { v8h v; _Float16 h[8]; } pk[4];
#pragma unroll
    for (int j = 0; j < 8; ++j)
#pragma unroll
      for (int i = 0; i < 4; ++i)
        pk[j >> 1].h[(j & 1) * 4 + i] = (_Float16)areg[j][i];
    _Float16* arow_p = &As[buf][arow * LSTRIDE + akoff];
#pragma unroll
    for (int j = 0; j < 4; ++j)
      *(v8h*)(arow_p + 8 * j) = pk[j].v;
    // B: transpose into [N][K] so fragment reads are contiguous b128
#pragma unroll
    for (int j = 0; j < 6; ++j)
#pragma unroll
      for (int i = 0; i < 4; ++i)
        Bs[buf][(bn0 + 4 * j + i) * LSTRIDE + bkk] = (_Float16)breg[j][i];
  };

  v8f acc[2][3];
  const v8f zero8 = {0.f, 0.f, 0.f, 0.f, 0.f, 0.f, 0.f, 0.f};
#pragma unroll
  for (int mt = 0; mt < 2; ++mt)
#pragma unroll
    for (int nt = 0; nt < 3; ++nt)
      acc[mt][nt] = zero8;

  // ---- prologue: fill buffer 0, start loads for buffer 1 ----
  load_regs(0);
  store_lds(0);
  if (NSTEP > 1) load_regs(BK);
  __syncthreads();

  for (int step = 0; step < NSTEP; ++step) {
    const int cur = step & 1;
    const _Float16* Ab = &As[cur][0];
    const _Float16* Bb = &Bs[cur][0];

    // ---- compute: two 32-K WMMA slabs out of the 64-K buffer ----
#pragma unroll
    for (int s = 0; s < 2; ++s) {
      // A (16-bit 16x32 layout): lane -> M = lane&15;
      //   element e -> K = 16*(e>>3) + 8*(lane>>4) + (e&7)
      FragAB afr[2];
#pragma unroll
      for (int mt = 0; mt < 2; ++mt) {
        const int base = (wm * 32 + mt * 16 + lm) * LSTRIDE + 32 * s;
        afr[mt].h[0] = *(const v8h*)&Ab[base + lg * 8];
        afr[mt].h[1] = *(const v8h*)&Ab[base + 16 + lg * 8];
      }
      // B (16-bit 32x16 layout): lane -> N = lane&15;
      //   element e -> K = 16*(e>>3) + 8*lg + (e&7)
      FragAB bfr[3];
#pragma unroll
      for (int nt = 0; nt < 3; ++nt) {
        const int base = (wn * 48 + nt * 16 + lm) * LSTRIDE + 32 * s;
        bfr[nt].h[0] = *(const v8h*)&Bb[base + lg * 8];
        bfr[nt].h[1] = *(const v8h*)&Bb[base + 16 + lg * 8];
      }
#pragma unroll
      for (int mt = 0; mt < 2; ++mt)
#pragma unroll
        for (int nt = 0; nt < 3; ++nt)
          acc[mt][nt] = __builtin_amdgcn_wmma_f32_16x16x32_f16(
              /*neg_a=*/false, afr[mt].v, /*neg_b=*/false, bfr[nt].v,
              /*c_mod=*/(short)0, acc[mt][nt],
              /*reuse_a=*/false, /*reuse_b=*/false);
    }

    // ---- stage next buffer, start loads for step+2, prefetch step+3 ----
    if (step + 1 < NSTEP) {
      store_lds(cur ^ 1);
      if (step + 2 < NSTEP) load_regs((step + 2) * BK);
      {
        const int kk = (step + 3) * BK + bkk;
        if (kk < EDGES)
          __builtin_prefetch(unroll_mat + ((size_t)b * EDGES + kk) * TARGETN +
                                 (Nbase + bn0), 0, 1);
      }
      // Single barrier per iteration: separates this iteration's stores to
      // buf[cur^1] from next iteration's reads; reads of buf[cur] already
      // retired (WMMA data dependence forced s_wait_dscnt before signal).
      __syncthreads();
    }
  }

  // ---- epilogue: scale by 1/occ (fp32) and store ----
  // C/D layout: VGPR r -> M = r + 8*(lane>>4), N = lane&15
  float rocc[3];
#pragma unroll
  for (int nt = 0; nt < 3; ++nt) {
    const int Ng = Nbase + wn * 48 + nt * 16 + lm;
    rocc[nt] = 1.0f / occ[(size_t)b * TARGETN + Ng];
  }
#pragma unroll
  for (int mt = 0; mt < 2; ++mt) {
#pragma unroll
    for (int nt = 0; nt < 3; ++nt) {
      const int Ng = Nbase + wn * 48 + nt * 16 + lm;
#pragma unroll
      for (int r = 0; r < 8; ++r) {
        const int Mg = Mbase + wm * 32 + mt * 16 + lg * 8 + r;
        out[((size_t)b * NFEAT + Mg) * TARGETN + Ng] = acc[mt][nt][r] * rocc[nt];
      }
    }
  }
}

extern "C" void kernel_launch(void* const* d_in, const int* in_sizes, int n_in,
                              void* d_out, int out_size, void* d_ws, size_t ws_size,
                              hipStream_t stream) {
  (void)in_sizes; (void)n_in; (void)out_size; (void)d_ws; (void)ws_size;
  const float* features   = (const float*)d_in[0];
  const float* unroll_mat = (const float*)d_in[1];
  const float* occ        = (const float*)d_in[2];
  float* out = (float*)d_out;

  dim3 grid(TARGETN / BN, NFEAT / BM, BATCH);   // (95, 2, 4)
  mesh_unpool_wmma<<<grid, dim3(256), 0, stream>>>(features, unroll_mat, occ, out);
}
